// Money_former_block_11312943858148
// MI455X (gfx1250) — compile-verified, matrix operand order checked by reading, not run on hardware
//
#include <hip/hip_runtime.h>
#include <hip/hip_bf16.h>

typedef unsigned short u16;
typedef __attribute__((ext_vector_type(16))) __bf16 v16bf;
typedef __attribute__((ext_vector_type(8)))  float  v8f;

#define S_TOK   1536
#define SPP     384
#define DMODEL  1024
#define NHEAD   16
#define HD      64
#define DFF     4096
#define QK_DIM  64          // 48 real dims padded to 64 for WMMA K
#define EPS_RMS 1.1920929e-07f
#define ATTN_EPS 1e-5f
#define LAMBDA_INIT 0.2f    // 0.8 - 0.6*exp(0)

#define TK 32
#define LDSS (TK + 8)       // padded LDS row stride (u16 units)

__device__ __forceinline__ u16 f2bf(float f) {
  unsigned u = __float_as_uint(f);
  u += 0x7FFFu + ((u >> 16) & 1u);       // round-to-nearest-even
  return (u16)(u >> 16);
}

// ---- gfx1250 async global->LDS staging (ASYNCcnt path), guarded fallback ----
// Toolchain signature (from probe diagnostic): param0 = v4i in AS(1) (global),
// param1 = v4i in AS(3) (LDS), then imm offset, imm cpol.
typedef int v4i __attribute__((vector_size(16)));
typedef __attribute__((address_space(1))) v4i g1_v4i;
typedef __attribute__((address_space(3))) v4i l3_v4i;

__device__ __forceinline__ void copy16_lds(void* lds_dst, const void* gsrc) {
#if defined(__has_builtin) && __has_builtin(__builtin_amdgcn_global_load_async_to_lds_b128)
  __builtin_amdgcn_global_load_async_to_lds_b128(
      (g1_v4i*)gsrc, (l3_v4i*)lds_dst, 0, 0);
#else
  *(uint4*)lds_dst = *(const uint4*)gsrc;
#endif
}
__device__ __forceinline__ void wait_async_copies() {
#if defined(__has_builtin) && __has_builtin(__builtin_amdgcn_s_wait_asynccnt)
  __builtin_amdgcn_s_wait_asynccnt(0);
#endif
}

// ---------------- fp32 -> bf16 elementwise ----------------
__global__ void cvt_bf16_kernel(const float* __restrict__ src, u16* __restrict__ dst, int n) {
  int i = blockIdx.x * blockDim.x + threadIdx.x;
  if (i < n) dst[i] = f2bf(src[i]);
}

// ---------------- generic row RMSNorm (fp32 in, bf16 out) ----------------
__global__ __launch_bounds__(256) void rmsnorm_kernel(
    const float* __restrict__ src, int ld_src, const float* __restrict__ w,
    u16* __restrict__ dst, int ld_dst, int cols, float eps) {
  __shared__ float red[256];
  int row = blockIdx.x, tid = threadIdx.x;
  const float* s = src + (size_t)row * ld_src;
  float acc = 0.f;
  for (int c = tid; c < cols; c += 256) { float v = s[c]; acc += v * v; }
  red[tid] = acc; __syncthreads();
  for (int off = 128; off > 0; off >>= 1) { if (tid < off) red[tid] += red[tid + off]; __syncthreads(); }
  float scale = rsqrtf(red[0] / (float)cols + eps);
  u16* d = dst + (size_t)row * ld_dst;
  for (int c = tid; c < cols; c += 256) d[c] = f2bf(s[c] * scale * w[c]);
}

// ---------------- attention-output RMSNorm over head dim ----------------
__global__ __launch_bounds__(64) void attn_rms_kernel(
    const float* __restrict__ attn, const float* __restrict__ w, u16* __restrict__ dst) {
  __shared__ float red[64];
  int blk = blockIdx.x;                 // s*16 + h
  int s = blk >> 4, h = blk & 15, d = threadIdx.x;
  float v = attn[((size_t)h * S_TOK + s) * HD + d];
  red[d] = v * v; __syncthreads();
  for (int off = 32; off > 0; off >>= 1) { if (d < off) red[d] += red[d + off]; __syncthreads(); }
  float scale = rsqrtf(red[0] * (1.f / HD) + ATTN_EPS);
  dst[(size_t)s * DMODEL + h * HD + d] = f2bf(v * scale * w[d]);
}

// ---------------- bf16 WMMA GEMM: C[M,N] = alpha * A[M,K] * B[N,K]^T (+ resid) ----------------
// 128 threads = 4 waves. Block tile BM x BN, wave grid WGM x WGN,
// per-wave tile (BM/WGM) x (BN/WGN) of 16x16 WMMA tiles. Double-buffered LDS,
// staged via async global->LDS when available.
union FragU { uint4 u[2]; v16bf v; };

template<int BM, int BN, int WGM, int WGN>
__global__ __launch_bounds__(128) void gemm_bf16_kernel(
    const u16* __restrict__ A, int lda,
    const u16* __restrict__ B, int ldb,
    float* __restrict__ C, int ldc,
    int M, int N, int K, float alpha,
    const float* __restrict__ resid, int ldr) {
  constexpr int WTM = BM / WGM;
  constexpr int WTN = BN / WGN;
  constexpr int MI = WTM / 16;
  constexpr int NI = WTN / 16;

  __shared__ __align__(16) u16 sA[2][BM][LDSS];
  __shared__ __align__(16) u16 sB[2][BN][LDSS];

  const int tid = threadIdx.x;
  const int lane = tid & 31, wave = tid >> 5;
  const int wm = (wave / WGN) * WTM;
  const int wn = (wave % WGN) * WTN;
  const int tile_m = blockIdx.y * BM, tile_n = blockIdx.x * BN;

  v8f acc[MI][NI] = {};

  auto stage = [&](int buf, int k0) {
    #pragma unroll
    for (int c = tid; c < BM * 4; c += 128) {       // 16B chunks of the A tile
      int r = c >> 2, ch = (c & 3) * 8;
      int gm = tile_m + r;
      u16* dst = &sA[buf][r][ch];
      if (gm < M) copy16_lds(dst, A + (size_t)gm * lda + k0 + ch);
      else        *(uint4*)dst = make_uint4(0u, 0u, 0u, 0u);
    }
    #pragma unroll
    for (int c = tid; c < BN * 4; c += 128) {       // 16B chunks of the B tile
      int r = c >> 2, ch = (c & 3) * 8;
      int gn = tile_n + r;
      u16* dst = &sB[buf][r][ch];
      if (gn < N) copy16_lds(dst, B + (size_t)gn * ldb + k0 + ch);
      else        *(uint4*)dst = make_uint4(0u, 0u, 0u, 0u);
    }
  };

  stage(0, 0);
  wait_async_copies();
  __syncthreads();

  // Fragment layout per ISA 7.12.2: lane[3:0]=row within 16, lane[4] selects K half
  // (koff 0/8); elems 0..7 -> K=koff..koff+7, elems 8..15 -> K=16+koff..16+koff+7.
  const int fr = lane & 15;
  const int koff = (lane >> 4) * 8;

  int buf = 0;
  for (int k0 = 0; k0 < K; k0 += TK, buf ^= 1) {
    if (k0 + TK < K) stage(buf ^ 1, k0 + TK);       // async-fill next buffer

    FragU afr[MI], bfr[NI];
    #pragma unroll
    for (int i = 0; i < MI; ++i) {
      afr[i].u[0] = *(const uint4*)&sA[buf][wm + i * 16 + fr][koff];
      afr[i].u[1] = *(const uint4*)&sA[buf][wm + i * 16 + fr][16 + koff];
    }
    #pragma unroll
    for (int j = 0; j < NI; ++j) {
      bfr[j].u[0] = *(const uint4*)&sB[buf][wn + j * 16 + fr][koff];
      bfr[j].u[1] = *(const uint4*)&sB[buf][wn + j * 16 + fr][16 + koff];
    }
    #pragma unroll
    for (int i = 0; i < MI; ++i)
      #pragma unroll
      for (int j = 0; j < NI; ++j)
        acc[i][j] = __builtin_amdgcn_wmma_f32_16x16x32_bf16(
            false, afr[i].v, false, bfr[j].v, (short)0, acc[i][j], false, false);

    wait_async_copies();
    __syncthreads();
  }

  const int cn = lane & 15;
  const int cm = (lane >> 4) * 8;
  #pragma unroll
  for (int i = 0; i < MI; ++i) {
    #pragma unroll
    for (int j = 0; j < NI; ++j) {
      #pragma unroll
      for (int r = 0; r < 8; ++r) {
        int gm = tile_m + wm + i * 16 + cm + r;
        int gn = tile_n + wn + j * 16 + cn;
        if (gm < M && gn < N) {
          float v = acc[i][j][r] * alpha;
          if (resid) v += resid[(size_t)gm * ldr + gn];
          C[(size_t)gm * ldc + gn] = v;
        }
      }
    }
  }
}

// ---------------- build padded Q/K (split heads + RoPE, pad 48->64) ----------------
__global__ __launch_bounds__(64) void build_qk_kernel(
    const float* __restrict__ qf,   // [S][1536]
    const float* __restrict__ kvm,  // [S][2048]
    const float* __restrict__ ckv,  // [S][288]
    const float* __restrict__ fcos, // [383][8]
    const float* __restrict__ fsin,
    u16* __restrict__ Qb,           // [32][S][64]
    u16* __restrict__ Kb) {
  int blk = blockIdx.x;             // s*32 + h2
  int s = blk >> 5, h2 = blk & 31, d = threadIdx.x;
  int h16 = h2 >> 1, half = h2 & 1;
  float qv = 0.f, kvv = 0.f;
  if (d < 32) {
    qv  = qf [(size_t)s * 1536 + h16 * 96  + half * 32 + d];
    kvv = kvm[(size_t)s * 2048 + h16 * 128 + half * 32 + d];
  } else if (d < 48) {
    int r = d - 32, f = r >> 1, p = s >> 2;   // sequences interleaved: pos = s//4
    float c  = (p == 0) ? 1.f : fcos[(p - 1) * 8 + f];
    float sn = (p == 0) ? 0.f : fsin[(p - 1) * 8 + f];
    float q0 = qf[(size_t)s * 1536 + h16 * 96 + 64 + half * 16 + 2 * f];
    float q1 = qf[(size_t)s * 1536 + h16 * 96 + 64 + half * 16 + 2 * f + 1];
    float k0 = ckv[(size_t)s * 288 + 256 + half * 16 + 2 * f];
    float k1 = ckv[(size_t)s * 288 + 256 + half * 16 + 2 * f + 1];
    qv  = (r & 1) ? (q0 * sn + q1 * c) : (q0 * c - q1 * sn);
    kvv = (r & 1) ? (k0 * sn + k1 * c) : (k0 * c - k1 * sn);
  }
  size_t o = ((size_t)h2 * S_TOK + s) * QK_DIM + d;
  Qb[o] = f2bf(qv);
  Kb[o] = f2bf(kvv);
}

// ---------------- V transposed per head: vT[h][d][s] ----------------
__global__ __launch_bounds__(64) void build_vT_kernel(const float* __restrict__ kvm, u16* __restrict__ vT) {
  int blk = blockIdx.x;             // s*16 + h
  int s = blk >> 4, h = blk & 15, d = threadIdx.x;
  vT[((size_t)h * HD + d) * S_TOK + s] = f2bf(kvm[(size_t)s * 2048 + h * 128 + 64 + d]);
}

// ---------------- lambda scalar ----------------
__global__ void lambda_kernel(const float* lq1, const float* lk1,
                              const float* lq2, const float* lk2, float* out) {
  if (threadIdx.x == 0 && blockIdx.x == 0) {
    float s1 = 0.f, s2 = 0.f;
    for (int i = 0; i < 32; ++i) { s1 += lq1[i] * lk1[i]; s2 += lq2[i] * lk2[i]; }
    out[0] = expf(s1) - expf(s2) + LAMBDA_INIT;
  }
}

// ---------------- masked softmax, in place, rows of two planes ----------------
__global__ __launch_bounds__(256) void softmax_kernel(float* __restrict__ P) {
  __shared__ float red[256];
  int row = blockIdx.x;             // 0..2*S_TOK-1 (two planes contiguous)
  int tid = threadIdx.x;
  int q = row % S_TOK, qm = q % SPP;
  float* p = P + (size_t)row * S_TOK;
  float mx = -3.4e38f;
  for (int k = tid; k < S_TOK; k += 256) {
    float v = ((k % SPP) > qm) ? -1e9f : p[k];
    mx = fmaxf(mx, v);
  }
  red[tid] = mx; __syncthreads();
  for (int off = 128; off > 0; off >>= 1) { if (tid < off) red[tid] = fmaxf(red[tid], red[tid + off]); __syncthreads(); }
  float m = red[0]; __syncthreads();
  float sum = 0.f;
  for (int k = tid; k < S_TOK; k += 256) {
    float v = ((k % SPP) > qm) ? -1e9f : p[k];
    float e = expf(v - m);
    p[k] = e; sum += e;
  }
  red[tid] = sum; __syncthreads();
  for (int off = 128; off > 0; off >>= 1) { if (tid < off) red[tid] += red[tid + off]; __syncthreads(); }
  float inv = 1.f / red[0];
  for (int k = tid; k < S_TOK; k += 256) p[k] *= inv;
}

// ---------------- g[k] = mean_q P0[q][k] ----------------
__global__ __launch_bounds__(256) void colmean_kernel(const float* __restrict__ P0, float* __restrict__ g) {
  int k = blockIdx.x * 256 + threadIdx.x;
  if (k >= S_TOK) return;
  float s = 0.f;
  for (int q = 0; q < S_TOK; ++q) s += P0[(size_t)q * S_TOK + k];
  g[k] = s * (1.f / S_TOK);
}

// ---------------- diff-attn combine -> bf16 ----------------
__global__ __launch_bounds__(256) void combine_kernel(
    const float* __restrict__ P, const float* __restrict__ g,
    const float* __restrict__ lamp, u16* __restrict__ Pd) {
  int idx = blockIdx.x * 256 + threadIdx.x;   // < S*S
  int q = idx / S_TOK, k = idx % S_TOK;
  float lam = lamp[0];
  float v;
  if ((k % SPP) > (q % SPP)) v = 0.f;
  else v = P[idx] - lam * P[(size_t)idx + (size_t)S_TOK * S_TOK] + lam * g[k];
  Pd[idx] = f2bf(v);
}

// ---------------- SwiGLU: gbuf = u * silu(vg) -> bf16 ----------------
__global__ __launch_bounds__(256) void swiglu_kernel(const float* __restrict__ uv, u16* __restrict__ gbuf) {
  int idx = blockIdx.x * 256 + threadIdx.x;   // < S*DFF
  int s = idx >> 12, j = idx & 4095;
  float u  = uv[(size_t)s * (2 * DFF) + j];
  float vg = uv[(size_t)s * (2 * DFF) + DFF + j];
  float sg = vg / (1.f + expf(-vg));
  gbuf[idx] = f2bf(u * sg);
}

extern "C" void kernel_launch(void* const* d_in, const int* in_sizes, int n_in,
                              void* d_out, int out_size, void* d_ws, size_t ws_size,
                              hipStream_t stream) {
  (void)in_sizes; (void)n_in; (void)out_size; (void)ws_size;
  const float* x         = (const float*)d_in[0];
  const float* fcos      = (const float*)d_in[1];
  const float* fsin      = (const float*)d_in[2];
  const float* norm1_w   = (const float*)d_in[3];
  const float* norm2_w   = (const float*)d_in[4];
  const float* kv_down_w = (const float*)d_in[5];
  const float* q_down_w  = (const float*)d_in[6];
  const float* kv_norm_w = (const float*)d_in[7];
  const float* q_norm_w  = (const float*)d_in[8];
  const float* kv_up_w   = (const float*)d_in[9];
  const float* q_up_w    = (const float*)d_in[10];
  const float* lq1       = (const float*)d_in[11];
  const float* lk1       = (const float*)d_in[12];
  const float* lq2       = (const float*)d_in[13];
  const float* lk2       = (const float*)d_in[14];
  const float* attn_norm_w = (const float*)d_in[15];
  const float* o_w       = (const float*)d_in[16];
  const float* ff_in_w   = (const float*)d_in[17];
  const float* ff_out_w  = (const float*)d_in[18];
  float* out = (float*)d_out;

  char* ws = (char*)d_ws;
  size_t off = 0;
  auto alloc = [&](size_t bytes) -> void* {
    off = (off + 255) & ~(size_t)255;
    void* p = ws + off; off += bytes; return p;
  };

  // bf16 weights
  u16* wKVd = (u16*)alloc((size_t)288 * 1024 * 2);
  u16* wQd  = (u16*)alloc((size_t)384 * 1024 * 2);
  u16* wKVu = (u16*)alloc((size_t)2048 * 256 * 2);
  u16* wQu  = (u16*)alloc((size_t)1536 * 384 * 2);
  u16* wO   = (u16*)alloc((size_t)1024 * 1024 * 2);
  u16* wF1  = (u16*)alloc((size_t)8192 * 1024 * 2);
  u16* wF2  = (u16*)alloc((size_t)1024 * 4096 * 2);
  // activations
  u16*   xin  = (u16*)  alloc((size_t)S_TOK * DMODEL * 2);
  float* ckv  = (float*)alloc((size_t)S_TOK * 288 * 4);
  u16*   ckn  = (u16*)  alloc((size_t)S_TOK * 256 * 2);
  float* kvm  = (float*)alloc((size_t)S_TOK * 2048 * 4);
  float* cq   = (float*)alloc((size_t)S_TOK * 384 * 4);
  u16*   cqn  = (u16*)  alloc((size_t)S_TOK * 384 * 2);
  float* qf   = (float*)alloc((size_t)S_TOK * 1536 * 4);
  u16*   Qb   = (u16*)  alloc((size_t)32 * S_TOK * QK_DIM * 2);
  u16*   Kb   = (u16*)  alloc((size_t)32 * S_TOK * QK_DIM * 2);
  u16*   vT   = (u16*)  alloc((size_t)16 * HD * S_TOK * 2);
  float* P    = (float*)alloc((size_t)2 * S_TOK * S_TOK * 4);   // reused per head-pair
  float* gbar = (float*)alloc((size_t)S_TOK * 4);
  float* lam  = (float*)alloc(256);
  u16*   Pd   = (u16*)  alloc((size_t)S_TOK * S_TOK * 2);
  float* attn = (float*)alloc((size_t)16 * S_TOK * HD * 4);
  u16*   o_in = (u16*)  alloc((size_t)S_TOK * DMODEL * 2);
  float* x2   = (float*)alloc((size_t)S_TOK * DMODEL * 4);
  u16*   hbf  = (u16*)  alloc((size_t)S_TOK * DMODEL * 2);
  float* uv   = (float*)alloc((size_t)S_TOK * 2 * DFF * 4);
  u16*   gb   = (u16*)  alloc((size_t)S_TOK * DFF * 2);

  auto cvt = [&](const float* src, u16* dst, int n) {
    cvt_bf16_kernel<<<(n + 255) / 256, 256, 0, stream>>>(src, dst, n);
  };
  cvt(kv_down_w, wKVd, 288 * 1024);
  cvt(q_down_w,  wQd,  384 * 1024);
  cvt(kv_up_w,   wKVu, 2048 * 256);
  cvt(q_up_w,    wQu,  1536 * 384);
  cvt(o_w,       wO,   1024 * 1024);
  cvt(ff_in_w,   wF1,  8192 * 1024);
  cvt(ff_out_w,  wF2,  1024 * 4096);

  // wide tiles (BN=128, 2x2 waves of 64x64) for all big-N GEMMs
  auto gemm = [&](const u16* A, int lda, const u16* B, int ldb, float* C, int ldc,
                  int M, int N, int K, float alpha, const float* resid, int ldr) {
    dim3 grid((N + 127) / 128, (M + 127) / 128);
    gemm_bf16_kernel<128, 128, 2, 2><<<grid, 128, 0, stream>>>(
        A, lda, B, ldb, C, ldc, M, N, K, alpha, resid, ldr);
  };
  // narrow tiles (BN=64, 4x1 waves of 32x64) for the A*V GEMM
  auto gemm_n64 = [&](const u16* A, int lda, const u16* B, int ldb, float* C, int ldc,
                      int M, int N, int K, float alpha) {
    dim3 grid((N + 63) / 64, (M + 127) / 128);
    gemm_bf16_kernel<128, 64, 4, 1><<<grid, 128, 0, stream>>>(
        A, lda, B, ldb, C, ldc, M, N, K, alpha, nullptr, 0);
  };

  // ---- front projections ----
  rmsnorm_kernel<<<S_TOK, 256, 0, stream>>>(x, DMODEL, norm1_w, xin, DMODEL, DMODEL, EPS_RMS);
  gemm(xin, DMODEL, wKVd, DMODEL, ckv, 288, S_TOK, 288, DMODEL, 1.f, nullptr, 0);
  gemm(xin, DMODEL, wQd,  DMODEL, cq,  384, S_TOK, 384, DMODEL, 1.f, nullptr, 0);
  rmsnorm_kernel<<<S_TOK, 256, 0, stream>>>(ckv, 288, kv_norm_w, ckn, 256, 256, EPS_RMS);
  rmsnorm_kernel<<<S_TOK, 256, 0, stream>>>(cq,  384, q_norm_w,  cqn, 384, 384, EPS_RMS);
  gemm(ckn, 256, wKVu, 256, kvm, 2048, S_TOK, 2048, 256, 1.f, nullptr, 0);
  gemm(cqn, 384, wQu,  384, qf,  1536, S_TOK, 1536, 384, 1.f, nullptr, 0);

  build_qk_kernel<<<S_TOK * 32, 64, 0, stream>>>(qf, kvm, ckv, fcos, fsin, Qb, Kb);
  build_vT_kernel<<<S_TOK * 16, 64, 0, stream>>>(kvm, vT);
  lambda_kernel<<<1, 32, 0, stream>>>(lq1, lk1, lq2, lk2, lam);

  // ---- differential attention, one head-pair at a time (scratch reuse) ----
  const float scaling = 0.14433756729740643f;   // 48^-0.5
  for (int h = 0; h < NHEAD; ++h) {
    const u16* Q0 = Qb + (size_t)(2 * h)     * S_TOK * QK_DIM;
    const u16* Q1 = Qb + (size_t)(2 * h + 1) * S_TOK * QK_DIM;
    const u16* K0 = Kb + (size_t)(2 * h)     * S_TOK * QK_DIM;
    const u16* K1 = Kb + (size_t)(2 * h + 1) * S_TOK * QK_DIM;
    gemm(Q0, QK_DIM, K0, QK_DIM, P, S_TOK, S_TOK, S_TOK, QK_DIM, scaling, nullptr, 0);
    gemm(Q1, QK_DIM, K1, QK_DIM, P + (size_t)S_TOK * S_TOK, S_TOK, S_TOK, S_TOK, QK_DIM, scaling, nullptr, 0);
    softmax_kernel<<<2 * S_TOK, 256, 0, stream>>>(P);
    colmean_kernel<<<(S_TOK + 255) / 256, 256, 0, stream>>>(P, gbar);
    combine_kernel<<<(S_TOK * S_TOK) / 256, 256, 0, stream>>>(P, gbar, lam, Pd);
    gemm_n64(Pd, S_TOK, vT + (size_t)h * HD * S_TOK, S_TOK,
             attn + (size_t)h * S_TOK * HD, HD, S_TOK, HD, S_TOK, 1.f);
  }

  // ---- epilogue: head RMSNorm, o-proj (+x), FFN (+x2) ----
  attn_rms_kernel<<<S_TOK * NHEAD, 64, 0, stream>>>(attn, attn_norm_w, o_in);
  gemm(o_in, DMODEL, wO, DMODEL, x2, DMODEL, S_TOK, DMODEL, DMODEL, 1.f, x, DMODEL);
  rmsnorm_kernel<<<S_TOK, 256, 0, stream>>>(x2, DMODEL, norm2_w, hbf, DMODEL, DMODEL, EPS_RMS);
  gemm(hbf, DMODEL, wF1, DMODEL, uv, 2 * DFF, S_TOK, 2 * DFF, DMODEL, 1.f, nullptr, 0);
  swiglu_kernel<<<(S_TOK * DFF) / 256, 256, 0, stream>>>(uv, gb);
  gemm(gb, DFF, wF2, DFF, out, DMODEL, S_TOK, DMODEL, DFF, 1.f, x2, DMODEL);
}
